// MemoryAugmentedMetaLearning_30236569764122
// MI455X (gfx1250) — matrix operand order repaired and checked by reading, compiled, not run on hardware
//
#include <hip/hip_runtime.h>
#include <hip/hip_bf16.h>

// ---------------------------------------------------------------------------
// MemoryAugmentedMetaLearning for gfx1250 (MI455X), wave32 + WMMA bf16 path.
// ---------------------------------------------------------------------------

typedef __attribute__((ext_vector_type(16))) __bf16 v16bf;
typedef __attribute__((ext_vector_type(8)))  __bf16 v8bf;
typedef __attribute__((ext_vector_type(8)))  float  v8f;

#define MEMN 1000
#define KEYD 64
#define VALD 128
#define DIND 512
#define BSUP 512
#define BQRY 65536

static __device__ __forceinline__ v8f vzero8() {
  v8f z;
#pragma unroll
  for (int i = 0; i < 8; ++i) z[i] = 0.f;
  return z;
}

static __device__ __forceinline__ v8f wmma_bf16(v16bf a, v16bf b, v8f c) {
  return __builtin_amdgcn_wmma_f32_16x16x32_bf16(false, a, false, b, (short)0, c, false, false);
}

// A-matrix (16x32 bf16) lane layout: lane r=lane&15 -> row M=r, h=lane>>4.
// elements 0..7  = K[h*8 .. h*8+7], elements 8..15 = K[16+h*8 .. 16+h*8+7].
static __device__ __forceinline__ v16bf load_a_bf16(const __bf16* src, int rowStride, int lane) {
  int r = lane & 15, h = lane >> 4;
  const __bf16* p0 = src + (size_t)r * rowStride + h * 8;
  const __bf16* p1 = p0 + 16;
  v16bf a;
#pragma unroll
  for (int i = 0; i < 8; ++i) { a[i] = p0[i]; a[8 + i] = p1[i]; }
  return a;
}

static __device__ __forceinline__ v16bf load_a_f32(const float* src, int rowStride, int lane) {
  int r = lane & 15, h = lane >> 4;
  const float* p0 = src + (size_t)r * rowStride + h * 8;
  const float* p1 = p0 + 16;
  v16bf a;
#pragma unroll
  for (int i = 0; i < 8; ++i) { a[i] = (__bf16)p0[i]; a[8 + i] = (__bf16)p1[i]; }
  return a;
}

// LDS A chunk: two aligned 16B halves (elements 0..7 at ap, 8..15 at ap+16).
static __device__ __forceinline__ v16bf load_a_lds(const __bf16* ap) {
  v8bf lo = *(const v8bf*)ap;
  v8bf hi = *(const v8bf*)(ap + 16);
  v16bf a;
#pragma unroll
  for (int i = 0; i < 8; ++i) { a[i] = lo[i]; a[8 + i] = hi[i]; }
  return a;
}

// Packed B tile: 512 bf16 per 16(K=32)x16(N) tile. Per-lane contiguous:
// element e (k = kt*32 + 8*h + e + (e >= 8 ? 8 : 0), n = nt*16 + (lane&15)) lives at
// dst[tile*512 + lane*16 + e]  ->  one 32B v16bf load = two global_load_b128.
static __device__ __forceinline__ v16bf load_b_tile(const __bf16* t, int lane) {
  return *(const v16bf*)(t + lane * 16);
}

// ---------------------------------------------------------------------------
// Weight packing: row-major f32 W[K][ld] -> packed bf16 B tiles (zero padded).
// ---------------------------------------------------------------------------
__global__ __launch_bounds__(256) void pack_w_kernel(const float* __restrict__ W, int ld,
                                                     int K, int N, __bf16* __restrict__ dst,
                                                     int nKt, int nNt) {
  int d = blockIdx.x * blockDim.x + threadIdx.x;
  int total = nKt * nNt * 256;
  if (d >= total) return;
  int tile = d >> 8, w = d & 255;
  int v = w >> 5, lane = w & 31;
  int kt = tile / nNt, nt = tile % nNt;
  int h = lane >> 4, n = nt * 16 + (lane & 15);
  int k0 = kt * 32 + 8 * h + 2 * v + (v >= 4 ? 8 : 0);
  float f0 = (k0 < K && n < N) ? W[(size_t)k0 * ld + n] : 0.f;
  float f1 = (k0 + 1 < K && n < N) ? W[(size_t)(k0 + 1) * ld + n] : 0.f;
  __bf16* tp = dst + (size_t)tile * 512 + lane * 16 + 2 * v;
  tp[0] = (__bf16)f0; tp[1] = (__bf16)f1;
}

// keys^T pack: B(k,n) = newKeys[n*64 + k]  (n = memory slot, zero for n>=1000)
__global__ __launch_bounds__(256) void pack_keys_kernel(const float* __restrict__ keys,
                                                        __bf16* __restrict__ dst) {
  int d = blockIdx.x * blockDim.x + threadIdx.x;
  if (d >= 2 * 64 * 256) return;
  int tile = d >> 8, w = d & 255;
  int v = w >> 5, lane = w & 31;
  int kt = tile / 64, nt = tile % 64;
  int h = lane >> 4, n = nt * 16 + (lane & 15);
  int k0 = kt * 32 + 8 * h + 2 * v + (v >= 4 ? 8 : 0);
  float f0 = 0.f, f1 = 0.f;
  if (n < MEMN) { f0 = keys[(size_t)n * KEYD + k0]; f1 = keys[(size_t)n * KEYD + k0 + 1]; }
  __bf16* tp = dst + (size_t)tile * 512 + lane * 16 + 2 * v;
  tp[0] = (__bf16)f0; tp[1] = (__bf16)f1;
}

// ---------------------------------------------------------------------------
// Support path: support_feat -> proj_keys, proj_vals  (tiny, scalar fp32)
// ---------------------------------------------------------------------------
__global__ __launch_bounds__(128) void support_kernel(const float* __restrict__ sx,
                                                      const float* __restrict__ sy,
                                                      const float* __restrict__ bw,
                                                      const float* __restrict__ bb,
                                                      const float* __restrict__ kw,
                                                      const float* __restrict__ kb,
                                                      const float* __restrict__ vw,
                                                      const float* __restrict__ vb,
                                                      float* __restrict__ projK,
                                                      float* __restrict__ projV) {
  __shared__ float lsx[DIND];
  __shared__ float lsy[VALD];
  __shared__ float lf[KEYD];
  int row = blockIdx.x, t = threadIdx.x;
  for (int i = t; i < DIND; i += 128) lsx[i] = sx[(size_t)row * DIND + i];
  lsy[t] = sy[(size_t)row * VALD + t];
  __syncthreads();
  if (t < KEYD) {
    float s = bb[t];
    for (int k = 0; k < DIND; ++k) s += lsx[k] * bw[(size_t)k * KEYD + t];
    lf[t] = s;
  }
  __syncthreads();
  if (t < KEYD) {
    float s = kb[t];
    for (int k = 0; k < KEYD; ++k) s += lf[k] * kw[(size_t)k * KEYD + t];
    projK[(size_t)row * KEYD + t] = s;
  }
  {
    float s = vb[t];
    for (int k = 0; k < VALD; ++k) s += lsy[k] * vw[(size_t)k * VALD + t];
    projV[(size_t)row * VALD + t] = s;
  }
}

// ---------------------------------------------------------------------------
// top_k(mem_age, 512) eviction + scatter (rank-select, jax tie semantics)
// ---------------------------------------------------------------------------
__global__ __launch_bounds__(128) void memwrite_kernel(const float* __restrict__ age,
                                                       const float* __restrict__ mk,
                                                       const float* __restrict__ mv,
                                                       const float* __restrict__ projK,
                                                       const float* __restrict__ projV,
                                                       float* __restrict__ nk,
                                                       float* __restrict__ nv) {
  __shared__ int scnt[128];
  int i = blockIdx.x, t = threadIdx.x;
  float ai = age[i];
  int c = 0;
  for (int j = t; j < MEMN; j += 128) {
    float aj = age[j];
    c += (aj > ai) || (aj == ai && j < i);
  }
  scnt[t] = c;
  __syncthreads();
  for (int s = 64; s > 0; s >>= 1) {
    if (t < s) scnt[t] += scnt[t + s];
    __syncthreads();
  }
  int rank = scnt[0];
  if (t < KEYD) {
    float v = mk[(size_t)i * KEYD + t];
    if (rank < BSUP) v = projK[(size_t)rank * KEYD + t];
    nk[(size_t)i * KEYD + t] = v;
  }
  {
    float v = mv[(size_t)i * VALD + t];
    if (rank < BSUP) v = projV[(size_t)rank * VALD + t];
    nv[(size_t)i * VALD + t] = v;
  }
}

// ---------------------------------------------------------------------------
// query_feat = query_x @ base_W + base_b   (WMMA bf16, software-pipelined)
// ---------------------------------------------------------------------------
__global__ __launch_bounds__(128) void qfeat_kernel(const float* __restrict__ qx,
                                                    const __bf16* __restrict__ pbase,
                                                    const float* __restrict__ bb,
                                                    __bf16* __restrict__ qfeat) {
  int lane = threadIdx.x & 31, wave = threadIdx.x >> 5;
  int rowG = blockIdx.x * 64 + wave * 16;
  int r = lane & 15, h = lane >> 4;
  v8f acc[4];
#pragma unroll
  for (int nt = 0; nt < 4; ++nt) acc[nt] = vzero8();

  v16bf a = load_a_f32(qx + (size_t)rowG * DIND, DIND, lane);
  v16bf bt[4];
#pragma unroll
  for (int nt = 0; nt < 4; ++nt) bt[nt] = load_b_tile(pbase + (size_t)nt * 512, lane);

#pragma unroll
  for (int kt = 0; kt < 16; ++kt) {
    v16bf an;
    v16bf btn[4];
    if (kt < 15) {                 // compile-time (loop fully unrolled)
      an = load_a_f32(qx + (size_t)rowG * DIND + (kt + 1) * 32, DIND, lane);
#pragma unroll
      for (int nt = 0; nt < 4; ++nt)
        btn[nt] = load_b_tile(pbase + (size_t)((kt + 1) * 4 + nt) * 512, lane);
    }
#pragma unroll
    for (int nt = 0; nt < 4; ++nt)
      acc[nt] = wmma_bf16(a, bt[nt], acc[nt]);
    if (kt < 15) {
      a = an;
#pragma unroll
      for (int nt = 0; nt < 4; ++nt) bt[nt] = btn[nt];
    }
  }
#pragma unroll
  for (int nt = 0; nt < 4; ++nt) {
    int col = nt * 16 + r;
    float bias = bb[col];
#pragma unroll
    for (int rr = 0; rr < 8; ++rr) {
      int row = rowG + rr + 8 * h;
      qfeat[(size_t)row * KEYD + col] = (__bf16)(acc[nt][rr] + bias);
    }
  }
}

// proj_q = query_feat @ kp_W + kp_b  (all tiles hoisted, then WMMA chain)
__global__ __launch_bounds__(128) void projq_kernel(const __bf16* __restrict__ qfeat,
                                                    const __bf16* __restrict__ pkp,
                                                    const float* __restrict__ kb,
                                                    __bf16* __restrict__ projq) {
  int lane = threadIdx.x & 31, wave = threadIdx.x >> 5;
  int rowG = blockIdx.x * 64 + wave * 16;
  int r = lane & 15, h = lane >> 4;
  v8f acc[4];
#pragma unroll
  for (int nt = 0; nt < 4; ++nt) acc[nt] = vzero8();

  v16bf a0 = load_a_bf16(qfeat + (size_t)rowG * KEYD, KEYD, lane);
  v16bf a1 = load_a_bf16(qfeat + (size_t)rowG * KEYD + 32, KEYD, lane);
  v16bf bt[8];
#pragma unroll
  for (int i = 0; i < 8; ++i) bt[i] = load_b_tile(pkp + (size_t)i * 512, lane);

#pragma unroll
  for (int nt = 0; nt < 4; ++nt) acc[nt] = wmma_bf16(a0, bt[nt], acc[nt]);
#pragma unroll
  for (int nt = 0; nt < 4; ++nt) acc[nt] = wmma_bf16(a1, bt[4 + nt], acc[nt]);

#pragma unroll
  for (int nt = 0; nt < 4; ++nt) {
    int col = nt * 16 + r;
    float bias = kb[col];
#pragma unroll
    for (int rr = 0; rr < 8; ++rr) {
      int row = rowG + rr + 8 * h;
      projq[(size_t)row * KEYD + col] = (__bf16)(acc[nt][rr] + bias);
    }
  }
}

// ---------------------------------------------------------------------------
// Fused attention: sims -> softmax -> retrieved  (64 query rows per block,
// sims kept in LDS, never touches HBM).  Row stride 1032 bf16 = 2064 B so the
// 16-byte A-chunk reads from 16 rows hit distinct LDS banks (516 % 64 == 4).
// ---------------------------------------------------------------------------
#define SSTR 1032
#define ATTN_LDS (64 * SSTR * 2 + 64 * 4 + 64 * 4 + 128 * 4)

__global__ __launch_bounds__(128) void attn_kernel(const __bf16* __restrict__ projq,
                                                   const __bf16* __restrict__ pkeys,
                                                   const __bf16* __restrict__ pvals,
                                                   float* __restrict__ retr) {
  extern __shared__ char smem[];
  __bf16* sS = (__bf16*)smem;
  float* sM = (float*)(smem + 64 * SSTR * 2);
  float* sL = sM + 64;
  float* sP = sL + 64;

  int lane = threadIdx.x & 31, wave = threadIdx.x >> 5;
  int rowG = blockIdx.x * 64 + wave * 16;
  int r = lane & 15, h = lane >> 4;

  // ---- pass 1: sims tile = proj_q(16x64) @ keys^T(64x1024) into LDS ----
  v16bf a0 = load_a_bf16(projq + (size_t)rowG * KEYD, KEYD, lane);
  v16bf a1 = load_a_bf16(projq + (size_t)rowG * KEYD + 32, KEYD, lane);
  v16bf b0c = load_b_tile(pkeys, lane);
  v16bf b1c = load_b_tile(pkeys + (size_t)64 * 512, lane);
#pragma unroll 4
  for (int nt = 0; nt < 64; ++nt) {
    int ntn = (nt + 1) & 63;   // wraps to 0 on last iter (harmless reload)
    v16bf b0n = load_b_tile(pkeys + (size_t)ntn * 512, lane);
    v16bf b1n = load_b_tile(pkeys + (size_t)(64 + ntn) * 512, lane);
    v8f acc = vzero8();
    acc = wmma_bf16(a0, b0c, acc);
    acc = wmma_bf16(a1, b1c, acc);
#pragma unroll
    for (int rr = 0; rr < 8; ++rr)
      sS[(size_t)(wave * 16 + rr + 8 * h) * SSTR + nt * 16 + r] = (__bf16)acc[rr];
    b0c = b0n;
    b1c = b1n;
  }
  __syncthreads();

  // ---- pass 2: per-row max / sum over valid cols [0,1000) ----
  int t = threadIdx.x, row = t >> 1, part = t & 1;
  int c0 = part * 500, c1 = c0 + 500;
  float m = -3.0e38f;
  for (int c = c0; c < c1; ++c) m = fmaxf(m, (float)sS[(size_t)row * SSTR + c]);
  sP[t] = m;
  __syncthreads();
  m = fmaxf(sP[row * 2], sP[row * 2 + 1]);
  __syncthreads();
  float l = 0.f;
  for (int c = c0; c < c1; ++c) l += __expf((float)sS[(size_t)row * SSTR + c] - m);
  sP[t] = l;
  __syncthreads();
  if (part == 0) {
    sM[row] = m;
    sL[row] = 1.f / (sP[row * 2] + sP[row * 2 + 1]);
  }
  __syncthreads();

  // ---- pass 3: overwrite scores with attn probabilities (bf16) ----
  m = sM[row];
  float il = sL[row];
  for (int c = c0; c < c1; ++c)
    sS[(size_t)row * SSTR + c] = (__bf16)(__expf((float)sS[(size_t)row * SSTR + c] - m) * il);
  if (part)
    for (int c = 1000; c < 1024; ++c) sS[(size_t)row * SSTR + c] = (__bf16)0.f;
  __syncthreads();

  // ---- pass 4: retrieved = attn(16x1024) @ values(1024x128), pipelined ----
  v8f acc[8];
#pragma unroll
  for (int nt = 0; nt < 8; ++nt) acc[nt] = vzero8();
  const __bf16* arow = sS + (size_t)(wave * 16 + r) * SSTR + h * 8;
  v16bf a = load_a_lds(arow);
  v16bf bc = load_b_tile(pvals, lane);
#pragma unroll 2
  for (int ks = 0; ks < 32; ++ks) {
    v16bf an = load_a_lds(arow + ((ks + 1) & 31) * 32);
#pragma unroll
    for (int nt = 0; nt < 8; ++nt) {
      int nidx = (ks * 8 + nt + 1) & 255;  // wraps on final iter
      v16bf bn = load_b_tile(pvals + (size_t)nidx * 512, lane);
      acc[nt] = wmma_bf16(a, bc, acc[nt]);
      bc = bn;
    }
    a = an;
  }
#pragma unroll
  for (int nt = 0; nt < 8; ++nt) {
    int col = nt * 16 + r;
#pragma unroll
    for (int rr = 0; rr < 8; ++rr)
      retr[(size_t)(rowG + rr + 8 * h) * VALD + col] = acc[nt][rr];
  }
}

// ---------------------------------------------------------------------------
// Fused controller MLP + loss:  relu(aug@c1) -> relu(@c2) -> @c3 -> MSE
// ---------------------------------------------------------------------------
__global__ __launch_bounds__(128) void mlp_kernel(const __bf16* __restrict__ qfeat,
                                                  const float* __restrict__ retr,
                                                  const __bf16* __restrict__ pc1,
                                                  const float* __restrict__ b1,
                                                  const __bf16* __restrict__ pc2,
                                                  const float* __restrict__ b2,
                                                  const float* __restrict__ c3w,
                                                  const float* __restrict__ c3b,
                                                  const float* __restrict__ qy,
                                                  float* __restrict__ out) {
  __shared__ __bf16 sH1[64 * 136];  // stride 272 B: 16B aligned, banks spread
  __shared__ __bf16 sH2[64 * 72];   // stride 144 B: 16B aligned
  int lane = threadIdx.x & 31, wave = threadIdx.x >> 5;
  int rowG = blockIdx.x * 64 + wave * 16;
  int r = lane & 15, h = lane >> 4;

  // h1 = relu([qfeat ; retrieved] @ c1 + b1),  K = 192, B prefetched 1 tile
  v8f acc[8];
#pragma unroll
  for (int nt = 0; nt < 8; ++nt) acc[nt] = vzero8();
  v16bf bc = load_b_tile(pc1, lane);
#pragma unroll
  for (int kt = 0; kt < 6; ++kt) {
    v16bf a = (kt < 2)
        ? load_a_bf16(qfeat + (size_t)rowG * KEYD + kt * 32, KEYD, lane)
        : load_a_f32(retr + (size_t)rowG * VALD + (kt - 2) * 32, VALD, lane);
#pragma unroll
    for (int nt = 0; nt < 8; ++nt) {
      int nidx = kt * 8 + nt + 1;
      v16bf bn = load_b_tile(pc1 + (size_t)(nidx < 48 ? nidx : 47) * 512, lane);
      acc[nt] = wmma_bf16(a, bc, acc[nt]);
      bc = bn;
    }
  }
#pragma unroll
  for (int nt = 0; nt < 8; ++nt) {
    int col = nt * 16 + r;
    float bias = b1[col];
#pragma unroll
    for (int rr = 0; rr < 8; ++rr)
      sH1[(size_t)(wave * 16 + rr + 8 * h) * 136 + col] = (__bf16)fmaxf(acc[nt][rr] + bias, 0.f);
  }
  __syncthreads();

  // h2 = relu(h1 @ c2 + b2),  K = 128
  v8f a2[4];
#pragma unroll
  for (int nt = 0; nt < 4; ++nt) a2[nt] = vzero8();
  v16bf b2c = load_b_tile(pc2, lane);
#pragma unroll
  for (int kt = 0; kt < 4; ++kt) {
    v16bf a = load_a_lds(sH1 + (size_t)(wave * 16 + r) * 136 + kt * 32 + h * 8);
#pragma unroll
    for (int nt = 0; nt < 4; ++nt) {
      int nidx = kt * 4 + nt + 1;
      v16bf bn = load_b_tile(pc2 + (size_t)(nidx < 16 ? nidx : 15) * 512, lane);
      a2[nt] = wmma_bf16(a, b2c, a2[nt]);
      b2c = bn;
    }
  }
#pragma unroll
  for (int nt = 0; nt < 4; ++nt) {
    int col = nt * 16 + r;
    float bias = b2[col];
#pragma unroll
    for (int rr = 0; rr < 8; ++rr)
      sH2[(size_t)(wave * 16 + rr + 8 * h) * 72 + col] = (__bf16)fmaxf(a2[nt][rr] + bias, 0.f);
  }
  __syncthreads();

  // pred = h2 @ c3 + b3 ;  loss += (pred - y)^2 / BQ
  float partial = 0.f;
  const __bf16* hp = sH2 + (size_t)(wave * 16 + r) * 72 + h * 32;
#pragma unroll
  for (int j = 0; j < 32; ++j) partial += (float)hp[j] * c3w[h * 32 + j];
  float other = __shfl_xor(partial, 16, 32);
  if (h == 0) {
    int row = rowG + r;
    float pred = partial + other + c3b[0];
    out[1 + row] = pred;
    float d = pred - qy[row];
    atomicAdd(out, d * d * (1.f / (float)BQRY));
  }
}

// ---------------------------------------------------------------------------
// Host-side orchestration
// ---------------------------------------------------------------------------
extern "C" void kernel_launch(void* const* d_in, const int* in_sizes, int n_in,
                              void* d_out, int out_size, void* d_ws, size_t ws_size,
                              hipStream_t stream) {
  (void)in_sizes; (void)n_in; (void)out_size; (void)ws_size;
  const float* support_x = (const float*)d_in[0];
  const float* support_y = (const float*)d_in[1];
  const float* query_x   = (const float*)d_in[2];
  const float* query_y   = (const float*)d_in[3];
  const float* base_W    = (const float*)d_in[4];
  const float* base_b    = (const float*)d_in[5];
  const float* kp_W      = (const float*)d_in[6];
  const float* kp_b      = (const float*)d_in[7];
  const float* vp_W      = (const float*)d_in[8];
  const float* vp_b      = (const float*)d_in[9];
  const float* mem_keys  = (const float*)d_in[10];
  const float* mem_vals  = (const float*)d_in[11];
  const float* mem_age   = (const float*)d_in[12];
  const float* c1_W      = (const float*)d_in[13];
  const float* c1_b      = (const float*)d_in[14];
  const float* c2_W      = (const float*)d_in[15];
  const float* c2_b      = (const float*)d_in[16];
  const float* c3_W      = (const float*)d_in[17];
  const float* c3_b      = (const float*)d_in[18];

  float* out = (float*)d_out;                 // [loss | query_pred | retrieved]
  float* retr = out + 1 + BQRY;               // [BQ,128]

  char* ws = (char*)d_ws;
  size_t off = 0;
  auto carve = [&](size_t bytes) { char* p = ws + off; off += (bytes + 255) & ~(size_t)255; return p; };

  float*  projK  = (float*)carve((size_t)BSUP * KEYD * 4);
  float*  projV  = (float*)carve((size_t)BSUP * VALD * 4);
  float*  newK   = (float*)carve((size_t)MEMN * KEYD * 4);
  float*  newV   = (float*)carve((size_t)MEMN * VALD * 4);
  __bf16* pbase  = (__bf16*)carve((size_t)16 * 4 * 512 * 2);
  __bf16* pkp    = (__bf16*)carve((size_t)2 * 4 * 512 * 2);
  __bf16* pc1    = (__bf16*)carve((size_t)6 * 8 * 512 * 2);
  __bf16* pc2    = (__bf16*)carve((size_t)4 * 4 * 512 * 2);
  __bf16* pkeys  = (__bf16*)carve((size_t)2 * 64 * 512 * 2);
  __bf16* pvals  = (__bf16*)carve((size_t)32 * 8 * 512 * 2);
  __bf16* qfeat  = (__bf16*)carve((size_t)BQRY * KEYD * 2);
  __bf16* projq  = (__bf16*)carve((size_t)BQRY * KEYD * 2);

  // static weight packing
  pack_w_kernel<<<(16 * 4 * 256 + 255) / 256, 256, 0, stream>>>(base_W, 64, 512, 64, pbase, 16, 4);
  pack_w_kernel<<<(2 * 4 * 256 + 255) / 256, 256, 0, stream>>>(kp_W, 64, 64, 64, pkp, 2, 4);
  pack_w_kernel<<<(6 * 8 * 256 + 255) / 256, 256, 0, stream>>>(c1_W, 128, 192, 128, pc1, 6, 8);
  pack_w_kernel<<<(4 * 4 * 256 + 255) / 256, 256, 0, stream>>>(c2_W, 64, 128, 64, pc2, 4, 4);

  // support path + memory write
  support_kernel<<<BSUP, 128, 0, stream>>>(support_x, support_y, base_W, base_b,
                                           kp_W, kp_b, vp_W, vp_b, projK, projV);
  memwrite_kernel<<<MEMN, 128, 0, stream>>>(mem_age, mem_keys, mem_vals, projK, projV, newK, newV);
  pack_keys_kernel<<<(2 * 64 * 256 + 255) / 256, 256, 0, stream>>>(newK, pkeys);
  pack_w_kernel<<<(32 * 8 * 256 + 255) / 256, 256, 0, stream>>>(newV, 128, MEMN, 128, pvals, 32, 8);

  // query path (heavy, WMMA)
  qfeat_kernel<<<BQRY / 64, 128, 0, stream>>>(query_x, pbase, base_b, qfeat);
  projq_kernel<<<BQRY / 64, 128, 0, stream>>>(qfeat, pkp, kp_b, projq);
  attn_kernel<<<BQRY / 64, 128, ATTN_LDS, stream>>>(projq, pkeys, pvals, retr);

  hipMemsetAsync(d_out, 0, sizeof(float), stream);
  mlp_kernel<<<BQRY / 64, 128, 0, stream>>>(qfeat, retr, pc1, c1_b, pc2, c2_b,
                                            c3_W, c3_b, query_y, out);
}
